// MyHandDecoder_39745627357288
// MI455X (gfx1250) — matrix operand order repaired and checked by reading, not compiled, not run on hardware
//
#include <hip/hip_runtime.h>
#include <math.h>

// ---------------------------------------------------------------------------
// MI455X (gfx1250) implementation of the MANO hand decoder.
// GEMMs: v_wmma_f32_16x16x32_bf16 with hi/lo bf16 split (3 WMMAs per K-step)
// for ~fp32 accuracy at bf16 matrix-pipe rate, fed by a double-buffered
// GLOBAL_LOAD_ASYNC_TO_LDS pipeline (ASYNCcnt split counter) so each block
// tile is pulled from L2 exactly once and the copy overlaps the WMMAs.
// ---------------------------------------------------------------------------

typedef __bf16 bf16;
typedef __attribute__((ext_vector_type(16))) __bf16 v16bf;
typedef __attribute__((ext_vector_type(8)))  __bf16 v8bf;
typedef __attribute__((ext_vector_type(8)))  float  v8f;

#define MESH   778
#define PD_K   135
#define PD_KP  160     // K padded to multiple of 32
#define PD_N   (MESH*3)
#define PD_NP  2432    // N padded to multiple of 128
#define NHEAD  640     // 5 concatenated 128-wide heads

// ------------------------------ conversion --------------------------------
__global__ void k_convert_hilo(const float* __restrict__ src,
                               bf16* __restrict__ hi, bf16* __restrict__ lo,
                               int rs, int cs, int rd, int cd)
{
  int idx = blockIdx.x * blockDim.x + threadIdx.x;
  if (idx >= rd * cd) return;
  int r = idx / cd, c = idx % cd;
  float x = (r < rs && c < cs) ? src[(size_t)r * cs + c] : 0.0f;
  bf16 h = (bf16)x;
  hi[idx] = h;
  lo[idx] = (bf16)(x - (float)h);
}

__global__ void k_copy_f32(const float* __restrict__ s, float* __restrict__ d, int n)
{
  int i = blockIdx.x * blockDim.x + threadIdx.x;
  if (i < n) d[i] = s[i];
}

// ------------------------------ WMMA GEMM ---------------------------------
// C(MxN) = A(MxK) * B(NxK)^T (+bias, optional relu).  A,B given as hi/lo bf16.

// one async 16B-per-lane copy global -> LDS (ASYNCcnt-tracked)
__device__ __forceinline__ void async_b128(unsigned lds_byte, const bf16* gbase,
                                           unsigned gbyte)
{
  asm volatile("global_load_async_to_lds_b128 %0, %1, %2"
               :: "v"(lds_byte), "v"(gbyte),
                  "s"((unsigned long long)(const void*)gbase)
               : "memory");
}

__device__ __forceinline__ v16bf lds_fragA(const bf16* p)
{
  v8bf c0 = *(const v8bf*)p;          // K chunk 0
  v8bf c1 = *(const v8bf*)(p + 16);   // K chunk +16
  return __builtin_shufflevector(c0, c1, 0,1,2,3,4,5,6,7,8,9,10,11,12,13,14,15);
}

__device__ __forceinline__ v16bf lds_fragB(const bf16* p)
{
  v8bf c0 = *(const v8bf*)p;          // K chunk 0
  v8bf c1 = *(const v8bf*)(p + 16);   // K chunk +16
  return __builtin_shufflevector(c0, c1, 0,1,2,3,4,5,6,7,8,9,10,11,12,13,14,15);
}

__device__ __forceinline__ v8f wmma_bf16(v16bf a, v16bf b, v8f c)
{
  return __builtin_amdgcn_wmma_f32_16x16x32_bf16(false, a, false, b,
                                                 (short)0, c, false, false);
}

// LDS layout (bf16 elements; rows padded to 40 elems = 80B = 20-bank stride):
//  Ah[buf] : buf*2560            Al[buf] : 5120  + buf*2560   (64x40 each)
//  Bh[buf] : 10240 + buf*5120    Bl[buf] : 20480 + buf*5120   (128x40 each)
__global__ __launch_bounds__(256) void k_gemm_bf16x2(
    const bf16* __restrict__ Ah, const bf16* __restrict__ Al,
    const bf16* __restrict__ Bh, const bf16* __restrict__ Bl,
    const float* __restrict__ bias, float* __restrict__ C,
    int M, int N, int K, int relu)
{
  __shared__ __align__(128) bf16 smem[30720];   // 60 KB, double buffered
  const int t    = threadIdx.x;
  const int lane = t & 31;
  const int wave = t >> 5;                      // 8 waves / block
  const int wm   = wave >> 2;
  const int wn   = wave & 3;
  const int m_blk = blockIdx.y * 64;
  const int n_blk = blockIdx.x * 128;
  // generic LDS address truncates to the 32-bit LDS byte offset (ISA 10.2)
  const unsigned smb = (unsigned)(size_t)&smem[0];

  const int rA = t >> 2;                        // 0..63 (A row / B half-row)
  const unsigned cb = (unsigned)(t & 3) << 4;   // 16B chunk within a row

  auto issue = [&](int k, int buf) {
    unsigned gA = ((unsigned)((m_blk + rA) * K + k) << 1) + cb;
    unsigned lA = (unsigned)rA * 80u + cb;
    async_b128(smb + (unsigned)buf * 5120u + lA,           Ah, gA);
    async_b128(smb + 10240u + (unsigned)buf * 5120u + lA,  Al, gA);
#pragma unroll
    for (int p = 0; p < 2; ++p) {
      int rB = (p << 6) + rA;                   // 0..127
      unsigned gB = ((unsigned)((n_blk + rB) * K + k) << 1) + cb;
      unsigned lB = (unsigned)rB * 80u + cb;
      async_b128(smb + 20480u + (unsigned)buf * 10240u + lB, Bh, gB);
      async_b128(smb + 40960u + (unsigned)buf * 10240u + lB, Bl, gB);
    }
    if (k + 64 < K) {                           // warm L2 two tiles ahead
      __builtin_prefetch(Ah + (size_t)(m_blk + rA) * K + k + 64, 0, 2);
      __builtin_prefetch(Bh + (size_t)(n_blk + rA) * K + k + 64, 0, 2);
    }
  };

  v8f acc[2][2] = {};
  int buf = 0;
  issue(0, 0);
  for (int k0 = 0; k0 < K; k0 += 32) {
    if (k0 + 32 < K) {
      issue(k0 + 32, buf ^ 1);                  // 6 more async ops in flight
      asm volatile("s_wait_asynccnt 0x6" ::: "memory");  // current tile done
    } else {
      asm volatile("s_wait_asynccnt 0x0" ::: "memory");
    }
    __syncthreads();                            // all waves' portions visible

    const unsigned aHe = (unsigned)buf * 2560u;
    const unsigned aLe = 5120u  + (unsigned)buf * 2560u;
    const unsigned bHe = 10240u + (unsigned)buf * 5120u;
    const unsigned bLe = 20480u + (unsigned)buf * 5120u;
    const int la  = lane & 15;
    const int ksA = (lane >> 4) << 3;           // A: chunks ks, ks+16
    const int ksB = (lane >> 4) << 3;           // B: chunks ks, ks+16

    v16bf ah[2], al[2], bh[2], bl[2];
#pragma unroll
    for (int mb = 0; mb < 2; ++mb) {
      int row = wm * 32 + mb * 16 + la;
      ah[mb] = lds_fragA(&smem[aHe + row * 40 + ksA]);
      al[mb] = lds_fragA(&smem[aLe + row * 40 + ksA]);
    }
#pragma unroll
    for (int nb = 0; nb < 2; ++nb) {
      int row = wn * 32 + nb * 16 + la;
      bh[nb] = lds_fragB(&smem[bHe + row * 40 + ksB]);
      bl[nb] = lds_fragB(&smem[bLe + row * 40 + ksB]);
    }
#pragma unroll
    for (int mb = 0; mb < 2; ++mb)
#pragma unroll
      for (int nb = 0; nb < 2; ++nb) {
        acc[mb][nb] = wmma_bf16(ah[mb], bh[nb], acc[mb][nb]);
        acc[mb][nb] = wmma_bf16(ah[mb], bl[nb], acc[mb][nb]);
        acc[mb][nb] = wmma_bf16(al[mb], bh[nb], acc[mb][nb]);
      }
    __syncthreads();                            // reads done before re-fill
    buf ^= 1;
  }

#pragma unroll
  for (int mb = 0; mb < 2; ++mb)
#pragma unroll
    for (int nb = 0; nb < 2; ++nb) {
      union { v8f v; float f[8]; } u; u.v = acc[mb][nb];
      int n     = n_blk + wn * 32 + nb * 16 + (lane & 15);
      int mbase = m_blk + wm * 32 + mb * 16 + ((lane >> 4) << 3);
      float bv  = bias ? bias[n] : 0.0f;
#pragma unroll
      for (int r = 0; r < 8; ++r) {
        float x = u.f[r] + bv;
        if (relu) x = fmaxf(x, 0.0f);
        C[(size_t)(mbase + r) * N + n] = x;
      }
    }
}

// ------------------------- batchnorm (deterministic) ----------------------
__global__ void k_colsum(const float* __restrict__ X, float* __restrict__ part,
                         int M, int N, int rows_per, int nchunk)
{
  int col = blockIdx.x * blockDim.x + threadIdx.x;
  int ch  = blockIdx.y;
  if (col >= N) return;
  int r0 = ch * rows_per, r1 = min(r0 + rows_per, M);
  float s = 0.f, q = 0.f;
  for (int r = r0; r < r1; ++r) { float x = X[(size_t)r * N + col]; s += x; q += x * x; }
  part[(size_t)ch * N + col]            = s;
  part[(size_t)(nchunk + ch) * N + col] = q;
}

__global__ void k_bnfin(const float* __restrict__ part, float* mean, float* rstd,
                        int N, int M, int nchunk)
{
  int col = blockIdx.x * blockDim.x + threadIdx.x;
  if (col >= N) return;
  float s = 0.f, q = 0.f;
  for (int c = 0; c < nchunk; ++c) {
    s += part[(size_t)c * N + col];
    q += part[(size_t)(nchunk + c) * N + col];
  }
  float m = s / (float)M;
  float v = q / (float)M - m * m;
  mean[col] = m;
  rstd[col] = rsqrtf(v + 1e-5f);
}

__global__ void k_bnapply(const float* __restrict__ X, const float* mean,
                          const float* rstd, const float* g, const float* bg,
                          bf16* __restrict__ hi, bf16* __restrict__ lo, int M, int N)
{
  int idx = blockIdx.x * blockDim.x + threadIdx.x;
  if (idx >= M * N) return;
  int n = idx % N;
  float y = (X[idx] - mean[n]) * rstd[n] * g[n] + bg[n];
  y = fmaxf(y, 0.0f);                       // relu fused
  bf16 h = (bf16)y;
  hi[idx] = h;
  lo[idx] = (bf16)(y - (float)h);
}

// --------------------------- small dense layers ---------------------------
__global__ void k_dense_small(const float* __restrict__ X, int ldx,
                              const float* __restrict__ W, const float* __restrict__ bias,
                              float* __restrict__ Y, int Bn, int N, int K,
                              int relu, int wtrans)
{
  int idx = blockIdx.x * blockDim.x + threadIdx.x;
  if (idx >= Bn * N) return;
  int b = idx / N, n = idx % N;
  const float* x = X + (size_t)b * ldx;
  float acc = bias ? bias[n] : 0.0f;
  if (wtrans) { for (int k = 0; k < K; ++k) acc += x[k] * W[(size_t)k * N + n]; }
  else        { for (int k = 0; k < K; ++k) acc += x[k] * W[(size_t)n * K + k]; }
  if (relu) acc = fmaxf(acc, 0.0f);
  Y[(size_t)b * N + n] = acc;
}

// ------------------------------- kinematics -------------------------------
__device__ void rodrigues9(float x, float y, float z, float* R)
{
  float t2 = x * x + y * y + z * z;
  float th = sqrtf(t2);
  float a, b, sx = x, sy = y, sz = z;
  if (th < 1e-30f) { a = 1.0f - t2 / 6.0f; b = 0.5f - t2 / 24.0f; }
  else { float inv = 1.0f / th; sx *= inv; sy *= inv; sz *= inv;
         a = sinf(th); b = 1.0f - cosf(th); }
  R[0] = 1.0f + b * (-(sy*sy + sz*sz));
  R[1] = -a * sz + b * (sx*sy);
  R[2] =  a * sy + b * (sx*sz);
  R[3] =  a * sz + b * (sx*sy);
  R[4] = 1.0f + b * (-(sx*sx + sz*sz));
  R[5] = -a * sx + b * (sy*sz);
  R[6] = -a * sy + b * (sx*sz);
  R[7] =  a * sx + b * (sy*sz);
  R[8] = 1.0f + b * (-(sx*sx + sy*sy));
}

// per-sample: joint rotations, pw (bf16 hi/lo, K padded to 160), final xform
__global__ void k_fk1(const float* __restrict__ poses, const float* __restrict__ rotv,
                      const float* __restrict__ scalev, const float* __restrict__ transv,
                      float* __restrict__ Rbuf, float* __restrict__ xform,
                      bf16* __restrict__ pwH, bf16* __restrict__ pwL, int Bn)
{
  int b = blockIdx.x * blockDim.x + threadIdx.x;
  if (b >= Bn) return;
  for (int j = 0; j < 16; ++j) {
    float x, y, z;
    if (j == 0) { x = 3.14159265358979323846f; y = 0.f; z = 0.f; }
    else { const float* p = poses + (size_t)b * 45 + (j - 1) * 3;
           x = p[0]; y = p[1]; z = p[2]; }
    float R[9];
    rodrigues9(x, y, z, R);
    float* Rd = Rbuf + (size_t)b * 144 + j * 9;
    for (int i = 0; i < 9; ++i) Rd[i] = R[i];
    if (j >= 1) {
      size_t base = (size_t)b * PD_KP + (j - 1) * 9;
      for (int i = 0; i < 9; ++i) {
        float v = R[i] - ((i % 4 == 0) ? 1.0f : 0.0f);
        bf16 h = (bf16)v;
        pwH[base + i] = h;
        pwL[base + i] = (bf16)(v - (float)h);
      }
    }
  }
  for (int c = PD_K; c < PD_KP; ++c) {
    pwH[(size_t)b * PD_KP + c] = (bf16)0.0f;
    pwL[(size_t)b * PD_KP + c] = (bf16)0.0f;
  }
  float R[9];
  rodrigues9(rotv[b * 3], rotv[b * 3 + 1], rotv[b * 3 + 2], R);
  float* xf = xform + (size_t)b * 16;
  for (int i = 0; i < 9; ++i) xf[i] = R[i];
  xf[9]  = fabsf(scalev[b]);
  xf[10] = transv[b * 3]; xf[11] = transv[b * 3 + 1]; xf[12] = transv[b * 3 + 2];
  xf[13] = xf[14] = xf[15] = 0.f;
}

// batch-independent: fold J_regressor through v_template and shapedirs
__global__ void k_jprep(const float* __restrict__ Jreg, const float* __restrict__ sd,
                        const float* __restrict__ vt, float* __restrict__ Jsd,
                        float* __restrict__ Jvt)
{
  int idx = blockIdx.x * blockDim.x + threadIdx.x;
  if (idx < 480) {
    int j = idx / 30, rem = idx % 30, c = rem / 10, k = rem % 10;
    float s = 0.f;
    for (int v = 0; v < MESH; ++v)
      s += Jreg[j * MESH + v] * sd[((size_t)v * 3 + c) * 10 + k];
    Jsd[idx] = s;
  } else if (idx < 528) {
    int i2 = idx - 480, j = i2 / 3, c = i2 % 3;
    float s = 0.f;
    for (int v = 0; v < MESH; ++v) s += Jreg[j * MESH + v] * vt[v * 3 + c];
    Jvt[i2] = s;
  }
}

__global__ void k_jcalc(const float* __restrict__ beta, const float* __restrict__ Jsd,
                        const float* __restrict__ Jvt, float* __restrict__ J, int Bn)
{
  int idx = blockIdx.x * blockDim.x + threadIdx.x;
  if (idx >= Bn * 48) return;
  int b = idx / 48, rem = idx % 48, j = rem / 3, c = rem % 3;
  float s = Jvt[rem];
  for (int k = 0; k < 10; ++k) s += beta[b * 10 + k] * Jsd[j * 30 + c * 10 + k];
  J[idx] = s;
}

// per-sample kinematic chain -> skinning matrices A (3x4 each) + joint trans
__global__ void k_fk2(const float* __restrict__ Rbuf, const float* __restrict__ J,
                      float* __restrict__ Apack, float* __restrict__ tj, int Bn)
{
  int b = blockIdx.x * blockDim.x + threadIdx.x;
  if (b >= Bn) return;
  const float* Rl = Rbuf + (size_t)b * 144;
  const float* Jl = J + (size_t)b * 48;
  float Rg[144], tg[48];
  for (int i = 0; i < 9; ++i) Rg[i] = Rl[i];
  tg[0] = Jl[0]; tg[1] = Jl[1]; tg[2] = Jl[2];
  const int PAR[16] = {-1,0,1,2,0,4,5,0,7,8,0,10,11,0,13,14};
#pragma unroll
  for (int j = 1; j < 16; ++j) {
    int p = PAR[j];
    const float* Rp = Rg + p * 9;
    const float* Rj = Rl + j * 9;
    float* Ro = Rg + j * 9;
#pragma unroll
    for (int r = 0; r < 3; ++r)
#pragma unroll
      for (int cc = 0; cc < 3; ++cc)
        Ro[r*3+cc] = Rp[r*3] * Rj[cc] + Rp[r*3+1] * Rj[3+cc] + Rp[r*3+2] * Rj[6+cc];
    float dx = Jl[j*3] - Jl[p*3], dy = Jl[j*3+1] - Jl[p*3+1], dz = Jl[j*3+2] - Jl[p*3+2];
#pragma unroll
    for (int r = 0; r < 3; ++r)
      tg[j*3+r] = Rp[r*3] * dx + Rp[r*3+1] * dy + Rp[r*3+2] * dz + tg[p*3+r];
  }
  float* Ao = Apack + (size_t)b * 192;
  float* to = tj + (size_t)b * 48;
  for (int j = 0; j < 16; ++j) {
    const float* Rj = Rg + j * 9;
    float jx = Jl[j*3], jy = Jl[j*3+1], jz = Jl[j*3+2];
    for (int r = 0; r < 3; ++r) {
      float u = Rj[r*3] * jx + Rj[r*3+1] * jy + Rj[r*3+2] * jz;
      Ao[j*12 + r*4 + 0] = Rj[r*3];
      Ao[j*12 + r*4 + 1] = Rj[r*3+1];
      Ao[j*12 + r*4 + 2] = Rj[r*3+2];
      Ao[j*12 + r*4 + 3] = tg[j*3+r] - u;
      to[j*3+r] = tg[j*3+r];
    }
  }
}

// 16 skeleton joint columns of the output (tips come from k_lbs)
__global__ void k_joints(const float* __restrict__ xform, const float* __restrict__ tj,
                         float* __restrict__ out, int Bn)
{
  int idx = blockIdx.x * blockDim.x + threadIdx.x;
  if (idx >= Bn * 48) return;
  int b = idx / 48, rem = idx % 48, j = rem / 3, i = rem % 3;
  const int CMAP[16] = {0,1,2,3,5,6,7,9,10,11,13,14,15,17,18,19};
  const float* xf = xform + (size_t)b * 16;
  const float* q  = tj + (size_t)b * 48 + j * 3;
  float p = xf[i*3] * q[0] + xf[i*3+1] * q[1] + xf[i*3+2] * q[2];
  out[(size_t)b * 63 + CMAP[j] * 3 + i] = xf[10 + i] + xf[9] * p;
}

// linear blend skinning + global rotation/scale/translation; one block per sample
__global__ __launch_bounds__(256) void k_lbs(
    const float* __restrict__ Apack, const float* __restrict__ xform,
    const float* __restrict__ beta, const float* __restrict__ vt,
    const float* __restrict__ sd, const float* __restrict__ wts,
    const float* __restrict__ P, float* __restrict__ out, int Bn)
{
  int b = blockIdx.x;
  int t = threadIdx.x;
  __shared__ float sA[192], sx[16], sb[10];
  if (t < 192)       sA[t]       = Apack[(size_t)b * 192 + t];
  else if (t < 208)  sx[t - 192] = xform[(size_t)b * 16 + (t - 192)];
  else if (t < 218)  sb[t - 208] = beta[(size_t)b * 10 + (t - 208)];
  __syncthreads();
  const size_t VOFF = (size_t)Bn * 63;
  for (int v = t; v < MESH; v += 256) {
    float rest[3];
    for (int c = 0; c < 3; ++c) {
      float r = vt[v * 3 + c] + P[(size_t)b * PD_NP + v * 3 + c];
      for (int k = 0; k < 10; ++k) r += sb[k] * sd[((size_t)v * 3 + c) * 10 + k];
      rest[c] = r;
    }
    float p0 = 0.f, p1 = 0.f, p2 = 0.f;
#pragma unroll
    for (int j = 0; j < 16; ++j) {
      float w = wts[(size_t)v * 16 + j];
      const float* Aj = sA + j * 12;
      p0 += w * (Aj[0]*rest[0] + Aj[1]*rest[1] + Aj[2] *rest[2] + Aj[3]);
      p1 += w * (Aj[4]*rest[0] + Aj[5]*rest[1] + Aj[6] *rest[2] + Aj[7]);
      p2 += w * (Aj[8]*rest[0] + Aj[9]*rest[1] + Aj[10]*rest[2] + Aj[11]);
    }
    float o[3];
    for (int i = 0; i < 3; ++i)
      o[i] = sx[10 + i] + sx[9] * (sx[i*3] * p0 + sx[i*3+1] * p1 + sx[i*3+2] * p2);
    float* vo = out + VOFF + ((size_t)b * MESH + v) * 3;
    vo[0] = o[0]; vo[1] = o[1]; vo[2] = o[2];
    int col = -1;
    if      (v == 320) col = 4;
    else if (v == 443) col = 8;
    else if (v == 672) col = 12;
    else if (v == 555) col = 16;
    else if (v == 744) col = 20;
    if (col >= 0) {
      float* jo = out + (size_t)b * 63 + col * 3;
      jo[0] = o[0]; jo[1] = o[1]; jo[2] = o[2];
    }
  }
}

// ------------------------------- host side --------------------------------
extern "C" void kernel_launch(void* const* d_in, const int* in_sizes, int n_in,
                              void* d_out, int out_size, void* d_ws, size_t ws_size,
                              hipStream_t stream)
{
  (void)n_in; (void)out_size; (void)ws_size;
  const float* features = (const float*)d_in[0];
  const float* W1  = (const float*)d_in[1];
  const float* b1  = (const float*)d_in[2];
  const float* g1  = (const float*)d_in[3];
  const float* bg1 = (const float*)d_in[4];
  const float* W2  = (const float*)d_in[5];
  const float* b2  = (const float*)d_in[6];
  const float* g2  = (const float*)d_in[7];
  const float* bg2 = (const float*)d_in[8];
  const float* headW1[5] = { (const float*)d_in[9],  (const float*)d_in[13],
                             (const float*)d_in[17], (const float*)d_in[23],
                             (const float*)d_in[29] };           // pose,shape,trans,rot,scale
  const float* headB1[5] = { (const float*)d_in[10], (const float*)d_in[14],
                             (const float*)d_in[18], (const float*)d_in[24],
                             (const float*)d_in[30] };
  const float* pose_W2  = (const float*)d_in[11];
  const float* pose_b2  = (const float*)d_in[12];
  const float* shape_W2 = (const float*)d_in[15];
  const float* shape_b2 = (const float*)d_in[16];
  const float* trans_W2 = (const float*)d_in[19];
  const float* trans_b2 = (const float*)d_in[20];
  const float* trans_W3 = (const float*)d_in[21];
  const float* trans_b3 = (const float*)d_in[22];
  const float* rot_W2   = (const float*)d_in[25];
  const float* rot_b2   = (const float*)d_in[26];
  const float* rot_W3   = (const float*)d_in[27];
  const float* rot_b3   = (const float*)d_in[28];
  const float* scale_W2 = (const float*)d_in[31];
  const float* scale_b2 = (const float*)d_in[32];
  const float* scale_W3 = (const float*)d_in[33];
  const float* scale_b3 = (const float*)d_in[34];
  const float* v_template = (const float*)d_in[35];
  const float* shapedirs  = (const float*)d_in[36];
  const float* posedirs   = (const float*)d_in[37];
  const float* J_reg      = (const float*)d_in[38];
  const float* weights    = (const float*)d_in[39];
  const float* hands_comp = (const float*)d_in[40];
  const float* hands_mean = (const float*)d_in[41];
  float* out = (float*)d_out;

  const int Bn = in_sizes[0] / 1536;

  // bump allocator on workspace (256B aligned)
  char* ws = (char*)d_ws;
  size_t off = 0;
  auto alloc = [&](size_t bytes) -> char* {
    off = (off + 255) & ~(size_t)255;
    char* p = ws + off;
    off += bytes;
    return p;
  };
  bf16*  featH = (bf16*)alloc((size_t)Bn * 1536 * 2);
  bf16*  featL = (bf16*)alloc((size_t)Bn * 1536 * 2);
  bf16*  w1H   = (bf16*)alloc((size_t)1024 * 1536 * 2);
  bf16*  w1L   = (bf16*)alloc((size_t)1024 * 1536 * 2);
  float* H1    = (float*)alloc((size_t)Bn * 1024 * 4);
  float* part1 = (float*)alloc((size_t)32 * 1024 * 4);
  float* mean1 = (float*)alloc(1024 * 4);
  float* rstd1 = (float*)alloc(1024 * 4);
  bf16*  h1H   = (bf16*)alloc((size_t)Bn * 1024 * 2);
  bf16*  h1L   = (bf16*)alloc((size_t)Bn * 1024 * 2);
  bf16*  w2H   = (bf16*)alloc((size_t)512 * 1024 * 2);
  bf16*  w2L   = (bf16*)alloc((size_t)512 * 1024 * 2);
  float* H2    = (float*)alloc((size_t)Bn * 512 * 4);
  float* part2 = (float*)alloc((size_t)32 * 512 * 4);
  float* mean2 = (float*)alloc(512 * 4);
  float* rstd2 = (float*)alloc(512 * 4);
  bf16*  h2H   = (bf16*)alloc((size_t)Bn * 512 * 2);
  bf16*  h2L   = (bf16*)alloc((size_t)Bn * 512 * 2);
  bf16*  hwH   = (bf16*)alloc((size_t)NHEAD * 512 * 2);
  bf16*  hwL   = (bf16*)alloc((size_t)NHEAD * 512 * 2);
  float* hb    = (float*)alloc(NHEAD * 4);
  float* Hh    = (float*)alloc((size_t)Bn * NHEAD * 4);
  float* theta = (float*)alloc((size_t)Bn * 30 * 4);
  float* betaB = (float*)alloc((size_t)Bn * 10 * 4);
  float* z2t   = (float*)alloc((size_t)Bn * 32 * 4);
  float* z2r   = (float*)alloc((size_t)Bn * 32 * 4);
  float* z2s   = (float*)alloc((size_t)Bn * 32 * 4);
  float* transB= (float*)alloc((size_t)Bn * 3 * 4);
  float* rotB  = (float*)alloc((size_t)Bn * 3 * 4);
  float* scaleB= (float*)alloc((size_t)Bn * 4);
  float* posesB= (float*)alloc((size_t)Bn * 45 * 4);
  float* Rbuf  = (float*)alloc((size_t)Bn * 144 * 4);
  float* xform = (float*)alloc((size_t)Bn * 16 * 4);
  bf16*  pwH   = (bf16*)alloc((size_t)Bn * PD_KP * 2);
  bf16*  pwL   = (bf16*)alloc((size_t)Bn * PD_KP * 2);
  bf16*  pdH   = (bf16*)alloc((size_t)PD_NP * PD_KP * 2);
  bf16*  pdL   = (bf16*)alloc((size_t)PD_NP * PD_KP * 2);
  float* Pbuf  = (float*)alloc((size_t)Bn * PD_NP * 4);
  float* Jsd   = (float*)alloc(480 * 4);
  float* Jvt   = (float*)alloc(48 * 4);
  float* Jbuf  = (float*)alloc((size_t)Bn * 48 * 4);
  float* Apack = (float*)alloc((size_t)Bn * 192 * 4);
  float* tj    = (float*)alloc((size_t)Bn * 48 * 4);

  const int TB = 256;
  auto cdiv = [](int a, int b) { return (a + b - 1) / b; };

  // ---- operand conversions (fp32 -> bf16 hi/lo) ----
  k_convert_hilo<<<cdiv(Bn * 1536, TB), TB, 0, stream>>>(features, featH, featL,
                                                         Bn, 1536, Bn, 1536);
  k_convert_hilo<<<cdiv(1024 * 1536, TB), TB, 0, stream>>>(W1, w1H, w1L,
                                                           1024, 1536, 1024, 1536);
  k_convert_hilo<<<cdiv(512 * 1024, TB), TB, 0, stream>>>(W2, w2H, w2L,
                                                          512, 1024, 512, 1024);
  for (int i = 0; i < 5; ++i) {
    k_convert_hilo<<<cdiv(128 * 512, TB), TB, 0, stream>>>(
        headW1[i], hwH + (size_t)i * 128 * 512, hwL + (size_t)i * 128 * 512,
        128, 512, 128, 512);
    k_copy_f32<<<1, 128, 0, stream>>>(headB1[i], hb + i * 128, 128);
  }
  k_convert_hilo<<<cdiv(PD_NP * PD_KP, TB), TB, 0, stream>>>(posedirs, pdH, pdL,
                                                             PD_N, PD_K, PD_NP, PD_KP);

  // ---- MLP trunk: GEMM -> BN(batch stats) + relu -> GEMM ... ----
  k_gemm_bf16x2<<<dim3(1024 / 128, Bn / 64), 256, 0, stream>>>(
      featH, featL, w1H, w1L, b1, H1, Bn, 1024, 1536, 0);
  k_colsum<<<dim3(cdiv(1024, TB), 16), TB, 0, stream>>>(H1, part1, Bn, 1024, Bn / 16, 16);
  k_bnfin<<<cdiv(1024, TB), TB, 0, stream>>>(part1, mean1, rstd1, 1024, Bn, 16);
  k_bnapply<<<cdiv(Bn * 1024, TB), TB, 0, stream>>>(H1, mean1, rstd1, g1, bg1,
                                                    h1H, h1L, Bn, 1024);

  k_gemm_bf16x2<<<dim3(512 / 128, Bn / 64), 256, 0, stream>>>(
      h1H, h1L, w2H, w2L, b2, H2, Bn, 512, 1024, 0);
  k_colsum<<<dim3(cdiv(512, TB), 16), TB, 0, stream>>>(H2, part2, Bn, 512, Bn / 16, 16);
  k_bnfin<<<cdiv(512, TB), TB, 0, stream>>>(part2, mean2, rstd2, 512, Bn, 16);
  k_bnapply<<<cdiv(Bn * 512, TB), TB, 0, stream>>>(H2, mean2, rstd2, g2, bg2,
                                                   h2H, h2L, Bn, 512);

  // ---- five heads, batched into one N=640 WMMA GEMM (relu fused) ----
  k_gemm_bf16x2<<<dim3(NHEAD / 128, Bn / 64), 256, 0, stream>>>(
      h2H, h2L, hwH, hwL, hb, Hh, Bn, NHEAD, 512, 1);

  // ---- small dense tails ----
  k_dense_small<<<cdiv(Bn * 30, TB), TB, 0, stream>>>(Hh + 0,   NHEAD, pose_W2,  pose_b2,  theta, Bn, 30, 128, 0, 0);
  k_dense_small<<<cdiv(Bn * 10, TB), TB, 0, stream>>>(Hh + 128, NHEAD, shape_W2, shape_b2, betaB, Bn, 10, 128, 0, 0);
  k_dense_small<<<cdiv(Bn * 32, TB), TB, 0, stream>>>(Hh + 256, NHEAD, trans_W2, trans_b2, z2t,   Bn, 32, 128, 0, 0);
  k_dense_small<<<cdiv(Bn * 32, TB), TB, 0, stream>>>(Hh + 384, NHEAD, rot_W2,   rot_b2,   z2r,   Bn, 32, 128, 0, 0);
  k_dense_small<<<cdiv(Bn * 32, TB), TB, 0, stream>>>(Hh + 512, NHEAD, scale_W2, scale_b2, z2s,   Bn, 32, 128, 0, 0);
  k_dense_small<<<cdiv(Bn * 3, TB), TB, 0, stream>>>(z2t, 32, trans_W3, trans_b3, transB, Bn, 3, 32, 0, 0);
  k_dense_small<<<cdiv(Bn * 3, TB), TB, 0, stream>>>(z2r, 32, rot_W3,   rot_b3,   rotB,   Bn, 3, 32, 0, 0);
  k_dense_small<<<cdiv(Bn * 1, TB), TB, 0, stream>>>(z2s, 32, scale_W3, scale_b3, scaleB, Bn, 1, 32, 0, 0);
  k_dense_small<<<cdiv(Bn * 45, TB), TB, 0, stream>>>(theta, 30, hands_comp, hands_mean, posesB, Bn, 45, 30, 0, 1);

  // ---- kinematics ----
  k_fk1<<<cdiv(Bn, 128), 128, 0, stream>>>(posesB, rotB, scaleB, transB,
                                           Rbuf, xform, pwH, pwL, Bn);
  k_jprep<<<cdiv(528, TB), TB, 0, stream>>>(J_reg, shapedirs, v_template, Jsd, Jvt);
  k_jcalc<<<cdiv(Bn * 48, TB), TB, 0, stream>>>(betaB, Jsd, Jvt, Jbuf, Bn);
  k_fk2<<<cdiv(Bn, 128), 128, 0, stream>>>(Rbuf, Jbuf, Apack, tj, Bn);

  // ---- posedirs correction GEMM: pw(4096x160) x posedirs^T(2432x160) ----
  k_gemm_bf16x2<<<dim3(PD_NP / 128, Bn / 64), 256, 0, stream>>>(
      pwH, pwL, pdH, pdL, (const float*)nullptr, Pbuf, Bn, PD_NP, PD_KP, 0);

  // ---- outputs: 16 skeleton joints + skinned vertices (+ 5 tip joints) ----
  k_joints<<<cdiv(Bn * 48, TB), TB, 0, stream>>>(xform, tj, out, Bn);
  k_lbs<<<Bn, 256, 0, stream>>>(Apack, xform, betaB, v_template, shapedirs,
                                weights, Pbuf, out, Bn);
}
